// ModConv2D_20675972563383
// MI455X (gfx1250) — matrix operand order repaired and checked by reading, not compiled, not run on hardware
//
#include <hip/hip_runtime.h>
#include <hip/hip_bf16.h>

// ---------------------------------------------------------------------------
// ModConv2D (StyleGAN2 modulated conv), b=16, ic=oc=128, 3x3, 128x128.
// Per-batch implicit GEMM  out[oc, h*w] = Wd[oc, ic*9] x im2col(x),
// M=128, K=1152, N=16384, on v_wmma_f32_16x16x32_bf16 (f32 accumulate).
// Register-blocked 2x2 tiles per wave: 4 WMMAs per 8 fragment half-loads.
// Staging: vectorized float4 interior + scalar halo; bf16 convert once here.
// HBM traffic ~ x(134MB)+out(134MB) = 268MB -> ~11.5us floor @ 23.3 TB/s.
// ---------------------------------------------------------------------------

typedef __attribute__((ext_vector_type(16))) __bf16 v16bf;
typedef __attribute__((ext_vector_type(8)))  __bf16 v8bf;
typedef __attribute__((ext_vector_type(8)))  float  v8f;

static constexpr int B_  = 16;
static constexpr int IC  = 128;
static constexpr int OC  = 128;
static constexpr int H_  = 128;
static constexpr int W_  = 128;
static constexpr int KS  = 3;
static constexpr int TN  = 64;          // pixel tile width per workgroup
static constexpr int LDSC = TN + 2;     // 66: w halo
static constexpr int LDSS = 136;        // padded ic stride (272B: 16B-aligned, bank-skewed)

// ---------------------------------------------------------------------------
// Kernel 1: modulate + demodulate weights, emit bf16 in [b][khw][oc][ic] order
// so A-fragments are two contiguous 16B loads per lane.
// ---------------------------------------------------------------------------
__global__ __launch_bounds__(128) void demod_kernel(const float* __restrict__ weight,
                                                    const float* __restrict__ y,
                                                    __bf16* __restrict__ wsW) {
    __shared__ float red[128];
    const int b  = blockIdx.x >> 7;
    const int oc = blockIdx.x & 127;
    const int ic = threadIdx.x;

    const float s = y[b * IC + ic] + 1.0f;
    const float* wp = weight + ((size_t)oc * IC + ic) * (KS * KS);

    float wv[9];
    float ss = 0.0f;
#pragma unroll
    for (int k = 0; k < 9; ++k) {
        wv[k] = wp[k] * s;
        ss += wv[k] * wv[k];
    }
    red[ic] = ss;
    __syncthreads();
#pragma unroll
    for (int off = 64; off > 0; off >>= 1) {
        if (ic < off) red[ic] += red[ic + off];
        __syncthreads();
    }
    const float d = rsqrtf(red[0] + 1e-8f);

    __bf16* wo = wsW + (size_t)b * 9 * OC * IC;
#pragma unroll
    for (int k = 0; k < 9; ++k)                      // ic-contiguous, coalesced
        wo[((size_t)k * OC + oc) * IC + ic] = (__bf16)(wv[k] * d);
}

// ---------------------------------------------------------------------------
// Fragment assembly: concatenate two 8-element halves (16B loads each).
// ---------------------------------------------------------------------------
static __device__ __forceinline__ v16bf cat8(v8bf lo, v8bf hi) {
    v16bf r;
#pragma unroll
    for (int e = 0; e < 8; ++e) { r[e] = lo[e]; r[e + 8] = hi[e]; }
    return r;
}

// ---------------------------------------------------------------------------
// Kernel 2: implicit-GEMM conv. Grid (w-tiles, h, b); 256 threads = 8 waves.
// WG tile: 128 oc x 64 px. Wave (oc-pair 0..3, px-pair 0..1) owns a 2x2 block
// of 16x16 tiles -> 4 accumulators, 4 WMMAs per (tap, ic-chunk) step.
// ---------------------------------------------------------------------------
__global__ __launch_bounds__(256, 1) void modconv_wmma(const float* __restrict__ x,
                                                       const __bf16* __restrict__ wsW,
                                                       float* __restrict__ out) {
    __shared__ __attribute__((aligned(32))) __bf16 xs[3 * LDSC * LDSS];  // 53.9 KB

    const int w0 = blockIdx.x * TN;
    const int h0 = blockIdx.y;
    const int b  = blockIdx.z;
    const float* xb = x + (size_t)b * IC * H_ * W_;

    // ---- stage interior: 3 rows x 128 ic x 16 float4 chunks (coalesced,
    //      16B-aligned since w0 is a multiple of 64). Power-of-two indexing. ----
    for (int idx = threadIdx.x; idx < 3 * IC * 16; idx += 256) {
        const int v4 = idx & 15;            // float4 chunk within row
        const int ic = (idx >> 4) & 127;
        const int r  = idx >> 11;           // 0..2
        const int hh = h0 - 1 + r;
        float4 f = make_float4(0.f, 0.f, 0.f, 0.f);
        if ((unsigned)hh < (unsigned)H_)
            f = *(const float4*)(xb + ((size_t)ic * H_ + hh) * W_ + w0 + v4 * 4);
        const int base = (r * LDSC + 1 + v4 * 4) * LDSS + ic;
        xs[base           ] = (__bf16)f.x;
        xs[base + LDSS    ] = (__bf16)f.y;
        xs[base + 2 * LDSS] = (__bf16)f.z;
        xs[base + 3 * LDSS] = (__bf16)f.w;
    }
    // ---- stage halo columns c=0 (ww=w0-1) and c=65 (ww=w0+64): 768 elems ----
    for (int idx = threadIdx.x; idx < 3 * IC * 2; idx += 256) {
        const int side = idx & 1;
        const int ic   = (idx >> 1) & 127;
        const int r    = idx >> 8;
        const int hh = h0 - 1 + r;
        const int ww = side ? (w0 + TN) : (w0 - 1);
        float v = 0.0f;
        if ((unsigned)hh < (unsigned)H_ && (unsigned)ww < (unsigned)W_)
            v = xb[((size_t)ic * H_ + hh) * W_ + ww];
        xs[(r * LDSC + (side ? (TN + 1) : 0)) * LDSS + ic] = (__bf16)v;
    }
    __syncthreads();

    const int lane = threadIdx.x & 31;
    const int wave = threadIdx.x >> 5;       // 0..7
    const int oc0  = (wave & 3) * 32;        // oc pair base
    const int n0   = (wave >> 2) * 32;       // pixel pair base within TN
    const int hi   = lane >> 4;              // lane half (ISA lane-16 split)
    const int lm   = lane & 15;

    v8f acc00 = {}, acc01 = {}, acc10 = {}, acc11 = {};   // [ocTile][pxTile]

    const __bf16* wb = wsW + (size_t)b * 9 * OC * IC;
    // A rows for this lane (A: one row per lane): oc0+lm and oc0+16+lm
    const __bf16* wrow0 = wb + (size_t)(oc0 + lm) * IC;
    const __bf16* wrow1 = wb + (size_t)(oc0 + 16 + lm) * IC;

#pragma unroll
    for (int khw = 0; khw < 9; ++khw) {
        const int kh = khw / 3, kw = khw % 3;          // compile-time after unroll
        const __bf16* wA = wrow0 + (size_t)khw * OC * IC;
        const __bf16* wB = wrow1 + (size_t)khw * OC * IC;
        // B columns for this lane: n = n0+lm and n0+16+lm
        const int lds0 = (kh * LDSC + (kw + n0 + lm)) * LDSS;
        const int lds1 = lds0 + 16 * LDSS;

#pragma unroll
        for (int icc = 0; icc < IC; icc += 32) {
            // A fragments (16x32): lane holds k = {hi*8+e, 16+hi*8+e}, e=0..7
            v16bf a0 = cat8(*(const v8bf*)(wA + icc + hi * 8),
                            *(const v8bf*)(wA + icc + hi * 8 + 16));
            v16bf a1 = cat8(*(const v8bf*)(wB + icc + hi * 8),
                            *(const v8bf*)(wB + icc + hi * 8 + 16));
            // B fragments (32x16): lane holds k = {hi*8+e, 16+hi*8+e}, e=0..7
            v16bf b0 = cat8(*(const v8bf*)(&xs[lds0 + icc + hi * 8]),
                            *(const v8bf*)(&xs[lds0 + icc + hi * 8 + 16]));
            v16bf b1 = cat8(*(const v8bf*)(&xs[lds1 + icc + hi * 8]),
                            *(const v8bf*)(&xs[lds1 + icc + hi * 8 + 16]));

            acc00 = __builtin_amdgcn_wmma_f32_16x16x32_bf16(false, a0, false, b0, (short)0, acc00, false, false);
            acc01 = __builtin_amdgcn_wmma_f32_16x16x32_bf16(false, a0, false, b1, (short)0, acc01, false, false);
            acc10 = __builtin_amdgcn_wmma_f32_16x16x32_bf16(false, a1, false, b0, (short)0, acc10, false, false);
            acc11 = __builtin_amdgcn_wmma_f32_16x16x32_bf16(false, a1, false, b1, (short)0, acc11, false, false);
        }
    }

    // ---- store: C/D layout -> oc = ocTile + hi*8 + v, col = base + lm ----
    float* ob = out + ((size_t)b * OC * H_ + h0) * W_;
    const int col0 = w0 + n0 + lm;
#pragma unroll
    for (int v = 0; v < 8; ++v) {
        const int m0 = oc0 + hi * 8 + v;
        const int m1 = m0 + 16;
        ob[(size_t)m0 * H_ * W_ + col0]      = acc00[v];
        ob[(size_t)m0 * H_ * W_ + col0 + 16] = acc01[v];
        ob[(size_t)m1 * H_ * W_ + col0]      = acc10[v];
        ob[(size_t)m1 * H_ * W_ + col0 + 16] = acc11[v];
    }
}

// ---------------------------------------------------------------------------
extern "C" void kernel_launch(void* const* d_in, const int* in_sizes, int n_in,
                              void* d_out, int out_size, void* d_ws, size_t ws_size,
                              hipStream_t stream) {
    (void)in_sizes; (void)n_in; (void)out_size; (void)ws_size;
    const float* x      = (const float*)d_in[0];   // (16,128,128,128) f32
    const float* y      = (const float*)d_in[1];   // (16,128) f32
    const float* weight = (const float*)d_in[2];   // (128,128,3,3) f32
    float* out = (float*)d_out;                    // (16,128,128,128) f32
    __bf16* wsW = (__bf16*)d_ws;                   // 16*9*128*128*2 B = 4.5 MB

    demod_kernel<<<dim3(B_ * OC), 128, 0, stream>>>(weight, y, wsW);
    modconv_wmma<<<dim3(W_ / TN, H_, B_), 256, 0, stream>>>(x, wsW, out);
}
